// MHA_GQA_43508018708803
// MI455X (gfx1250) — matrix operand order repaired and checked by reading, not compiled, not run on hardware
//
#include <hip/hip_runtime.h>

// ---------------------------------------------------------------------------
// CDNA5 (gfx1250) GQA attention block: QKV proj -> RoPE -> flash attention
// -> out proj. All GEMMs via v_wmma_f32_16x16x32_bf16 (wave32, f32 accum).
// Global->LDS staging via GLOBAL_LOAD_ASYNC_TO_LDS_B128 (ASYNCcnt) with
// double-buffered LDS tiles.
// ---------------------------------------------------------------------------

typedef __attribute__((ext_vector_type(16))) __bf16 v16bf;
typedef __attribute__((ext_vector_type(8)))  __bf16 v8bf;
typedef __attribute__((ext_vector_type(8)))  float  v8f;

union FragBF {
  v16bf v;
  v8bf  h[2];
};

#ifndef __has_builtin
#define __has_builtin(x) 0
#endif

#if __has_builtin(__builtin_amdgcn_global_load_async_to_lds_b128)
#define HAVE_ASYNC_LDS 1
#else
#define HAVE_ASYNC_LDS 0
#endif

#if __has_builtin(__builtin_amdgcn_s_wait_asynccnt)
#define HAVE_WAIT_ASYNC 1
#else
#define HAVE_WAIT_ASYNC 0
#endif

#if HAVE_ASYNC_LDS
typedef int v4i __attribute__((vector_size(16)));
typedef __attribute__((address_space(1))) v4i g_v4i;  // global
typedef __attribute__((address_space(3))) v4i l_v4i;  // LDS
#endif

// 16B global -> LDS copy; async (no VGPR data path) when available.
__device__ __forceinline__ void async_copy_b128(void* lds, const void* g) {
#if HAVE_ASYNC_LDS
  __builtin_amdgcn_global_load_async_to_lds_b128((g_v4i*)g, (l_v4i*)lds, 0, 0);
#else
  *(uint4*)lds = *(const uint4*)g;
#endif
}

__device__ __forceinline__ void wait_async() {
#if HAVE_ASYNC_LDS
#if HAVE_WAIT_ASYNC
  __builtin_amdgcn_s_wait_asynccnt(0);
#else
  asm volatile("s_wait_asynccnt 0" ::: "memory");
#endif
#endif
}

__device__ __forceinline__ __bf16 f2bf(float f) {
  unsigned u = __builtin_bit_cast(unsigned, f);
  u += 0x7FFFu + ((u >> 16) & 1u);               // round-to-nearest-even
  unsigned short hs = (unsigned short)(u >> 16);
  return __builtin_bit_cast(__bf16, hs);
}

static constexpr int T_   = 2048;
static constexpr int DM_  = 2048;
static constexpr int NH_  = 32;
static constexpr int NKV_ = 8;
static constexpr int HD_  = 64;

// ---------------------------------------------------------------------------
// Elementwise f32 -> bf16
// ---------------------------------------------------------------------------
__global__ __launch_bounds__(256)
void cvt_f32_bf16_kernel(const float* __restrict__ in, __bf16* __restrict__ out, int n) {
  int i = blockIdx.x * 256 + threadIdx.x;
  if (i < n) out[i] = f2bf(in[i]);
}

// ---------------------------------------------------------------------------
// W[K][N] f32  ->  WT[N][K] bf16   (32x32 LDS tile transpose)
// ---------------------------------------------------------------------------
__global__ __launch_bounds__(256)
void transpose_f32_bf16_kernel(const float* __restrict__ W, __bf16* __restrict__ WT,
                               int K, int N) {
  __shared__ float tile[32][33];
  int bx = blockIdx.x * 32;  // N base
  int by = blockIdx.y * 32;  // K base
  int x = threadIdx.x;       // 0..31
  int y = threadIdx.y;       // 0..7
#pragma unroll
  for (int i = 0; i < 32; i += 8)
    tile[y + i][x] = W[(size_t)(by + y + i) * N + (bx + x)];
  __syncthreads();
#pragma unroll
  for (int i = 0; i < 32; i += 8)
    WT[(size_t)(bx + y + i) * K + (by + x)] = f2bf(tile[x][y + i]);
}

// ---------------------------------------------------------------------------
// C[M][N] f32 = A[M][K] bf16 @ BT[N][K] bf16.
// Block tile 128x128, 4 waves, wave tile 64x64 = 16 accumulators.
// K-step 64, double-buffered LDS, async global->LDS staging.
// ---------------------------------------------------------------------------
__global__ __launch_bounds__(128)
void gemm_bf16_nt_kernel(const __bf16* __restrict__ A,
                         const __bf16* __restrict__ BT,
                         float* __restrict__ C,
                         int M, int N, int K) {
  __shared__ __align__(16) __bf16 As[2][128][72];   // 64 K + 8 pad
  __shared__ __align__(16) __bf16 Bs[2][128][72];

  const int tid  = threadIdx.x;
  const int lane = tid & 31;
  const int w    = tid >> 5;
  const int wm   = (w >> 1) * 64;
  const int wn   = (w & 1) * 64;
  const int m0   = blockIdx.y * 128;
  const int n0   = blockIdx.x * 128;
  const int hh   = lane >> 4;       // lane half (ISA A/B lane layout)
  const int l16  = lane & 15;
  const int ab   = hh ? 8 : 0;      // unused; A frag also uses K chunk {bb..bb+15}
  const int bb   = hh ? 16 : 0;     // B frag: K chunk  {bb..bb+15}

  v8f acc[4][4];
#pragma unroll
  for (int i = 0; i < 4; ++i)
#pragma unroll
    for (int j = 0; j < 4; ++j)
#pragma unroll
      for (int r = 0; r < 8; ++r) acc[i][j][r] = 0.0f;

  const __bf16* aRow = A  + (size_t)(m0 + tid) * K;
  const __bf16* bRow = BT + (size_t)(n0 + tid) * K;

  auto issue = [&](int k0, int b) {
#pragma unroll
    for (int i = 0; i < 8; ++i)
      async_copy_b128(&As[b][tid][i * 8], aRow + k0 + i * 8);
#pragma unroll
    for (int i = 0; i < 8; ++i)
      async_copy_b128(&Bs[b][tid][i * 8], bRow + k0 + i * 8);
  };

  issue(0, 0);
  const int ntiles = K >> 6;
  for (int t = 0; t < ntiles; ++t) {
    const int cur = t & 1;
    wait_async();          // our async copies for buf[cur] complete
    __syncthreads();       // everyone's copies visible; buf[cur^1] free
    if (t + 1 < ntiles) issue((t + 1) << 6, cur ^ 1);

#pragma unroll
    for (int ks = 0; ks < 2; ++ks) {
      FragBF af[4], bfb[4];
#pragma unroll
      for (int mi = 0; mi < 4; ++mi) {
        const __bf16* p = &As[cur][wm + mi * 16 + l16][ks * 32 + bb];
        af[mi].h[0] = *(const v8bf*)p;
        af[mi].h[1] = *(const v8bf*)(p + 8);
      }
#pragma unroll
      for (int ni = 0; ni < 4; ++ni) {
        const __bf16* p = &Bs[cur][wn + ni * 16 + l16][ks * 32 + bb];
        bfb[ni].h[0] = *(const v8bf*)p;
        bfb[ni].h[1] = *(const v8bf*)(p + 8);
      }
#pragma unroll
      for (int mi = 0; mi < 4; ++mi)
#pragma unroll
        for (int ni = 0; ni < 4; ++ni)
          acc[mi][ni] = __builtin_amdgcn_wmma_f32_16x16x32_bf16(
              false, af[mi].v, false, bfb[ni].v, (short)0, acc[mi][ni], false, false);
    }
  }

#pragma unroll
  for (int mi = 0; mi < 4; ++mi)
#pragma unroll
    for (int ni = 0; ni < 4; ++ni)
#pragma unroll
      for (int r = 0; r < 8; ++r) {
        int m = m0 + wm + mi * 16 + r + hh * 8;
        int n = n0 + wn + ni * 16 + l16;
        C[(size_t)m * N + n] = acc[mi][ni][r];
      }
}

// ---------------------------------------------------------------------------
// RoPE + repack: in f32 [T][nH*64]  ->  out bf16 [nH][T][64]
// ---------------------------------------------------------------------------
__global__ __launch_bounds__(256)
void rope_pack_kernel(const float* __restrict__ in, __bf16* __restrict__ out,
                      int T, int nH) {
  int i = blockIdx.x * 256 + threadIdx.x;
  if (i >= nH * T * 32) return;
  int j = i & 31;
  int t = (i >> 5) % T;
  int h = i / (T * 32);
  float invf = __expf(-(float)j * (9.210340371976184f / 32.0f)); // 10000^(-j/32)
  float ang = (float)t * invf;
  float s, c;
  __sincosf(ang, &s, &c);
  const float* row = in + (size_t)t * (nH * 64) + h * 64;
  float x1 = row[j], x2 = row[j + 32];
  __bf16* orow = out + ((size_t)h * T + t) * 64;
  orow[j]      = f2bf(x1 * c - x2 * s);
  orow[j + 32] = f2bf(x2 * c + x1 * s);
}

// ---------------------------------------------------------------------------
// V pack: vf32 [T][NKV*64] -> Vt bf16 [NKV][64][T] (transposed, B-operand ready)
// ---------------------------------------------------------------------------
__global__ __launch_bounds__(256)
void pack_vT_kernel(const float* __restrict__ vf, __bf16* __restrict__ Vt, int T) {
  int i = blockIdx.x * 256 + threadIdx.x;
  if (i >= NKV_ * HD_ * T) return;
  int t   = i % T;
  int d   = (i / T) % HD_;
  int kvh = i / (T * HD_);
  Vt[i] = f2bf(vf[(size_t)t * (NKV_ * HD_) + kvh * HD_ + d]);
}

// ---------------------------------------------------------------------------
// Flash attention: block = (64 queries, 1 head); 4 waves x 16 query rows.
// Double-buffered K/V/mask tiles staged with async copies. Per 128-key tile:
// 16 WMMA for S = Q*K^T, register online softmax, P via per-wave LDS
// (C-layout -> A-layout), 16 WMMA for P*V.
// ---------------------------------------------------------------------------
__global__ __launch_bounds__(128)
void flash_attn_kernel(const __bf16* __restrict__ Qb,  // [NH][T][64]
                       const __bf16* __restrict__ Kb,  // [NKV][T][64]
                       const __bf16* __restrict__ Vt,  // [NKV][64][T]
                       const float* __restrict__ mask, // [T]
                       __bf16* __restrict__ Y,         // [T][NH*64]
                       int T) {
  __shared__ __align__(16) __bf16 Ks[2][128][72];     // keys x d (+8 pad)
  __shared__ __align__(16) __bf16 Vs[2][64][136];     // d x keys (+8 pad)
  __shared__ __align__(16) __bf16 Ps[4][16][136];     // per-wave P staging
  __shared__ float Ms[2][128];

  const int tid  = threadIdx.x;
  const int lane = tid & 31;
  const int w    = tid >> 5;
  const int h    = blockIdx.y;
  const int kvh  = h >> 2;               // GROUPS = 4
  const int q0   = blockIdx.x * 64;
  const int hh   = lane >> 4;
  const int l16  = lane & 15;
  const int ab   = hh ? 8 : 0;
  const int bb   = hh ? 16 : 0;

  // Q fragments for this wave's 16 query rows (d-chunks [0,32) and [32,64))
  FragBF aq[2];
  {
    const __bf16* qrow = Qb + ((size_t)h * T + q0 + w * 16 + l16) * 64;
    aq[0].h[0] = *(const v8bf*)(qrow + bb);
    aq[0].h[1] = *(const v8bf*)(qrow + bb + 8);
    aq[1].h[0] = *(const v8bf*)(qrow + 32 + bb);
    aq[1].h[1] = *(const v8bf*)(qrow + 32 + bb + 8);
  }

  float rm[8], rl[8];
  v8f o[4];
#pragma unroll
  for (int r = 0; r < 8; ++r) { rm[r] = -1.0e30f; rl[r] = 0.0f; }
#pragma unroll
  for (int di = 0; di < 4; ++di)
#pragma unroll
    for (int r = 0; r < 8; ++r) o[di][r] = 0.0f;

  auto issueKV = [&](int kt, int b) {
    const __bf16* gk = Kb + ((size_t)kvh * T + kt + tid) * 64;
#pragma unroll
    for (int i = 0; i < 4; ++i)
      async_copy_b128(&Ks[b][tid][i * 8], gk + i * 8);
    int d = tid >> 1, hc = (tid & 1) * 64;
    const __bf16* gv = Vt + ((size_t)kvh * 64 + d) * T + kt + hc;
#pragma unroll
    for (int i = 0; i < 8; ++i)
      async_copy_b128(&Vs[b][d][hc + i * 8], gv + i * 8);
    Ms[b][tid] = mask[kt + tid];
  };

  issueKV(0, 0);
  const int ntiles = T >> 7;  // 128 keys per tile
  for (int it = 0; it < ntiles; ++it) {
    const int cur = it & 1;
    wait_async();
    __syncthreads();
    if (it + 1 < ntiles) issueKV((it + 1) << 7, cur ^ 1);

    // --- S = Q * K^T (scaled) + additive mask ---
    v8f s[8];
#pragma unroll
    for (int nk = 0; nk < 8; ++nk) {
      FragBF b0, b1;
      const __bf16* kp = &Ks[cur][nk * 16 + l16][0];
      b0.h[0] = *(const v8bf*)(kp + bb);
      b0.h[1] = *(const v8bf*)(kp + bb + 8);
      b1.h[0] = *(const v8bf*)(kp + 32 + bb);
      b1.h[1] = *(const v8bf*)(kp + 32 + bb + 8);
      v8f z;
#pragma unroll
      for (int r = 0; r < 8; ++r) z[r] = 0.0f;
      z = __builtin_amdgcn_wmma_f32_16x16x32_bf16(false, aq[0].v, false, b0.v,
                                                  (short)0, z, false, false);
      s[nk] = __builtin_amdgcn_wmma_f32_16x16x32_bf16(false, aq[1].v, false, b1.v,
                                                      (short)0, z, false, false);
      float am = (1.0f - Ms[cur][nk * 16 + l16]) * -10000.0f;
#pragma unroll
      for (int r = 0; r < 8; ++r) s[nk][r] = s[nk][r] * 0.125f + am;
    }

    // --- online softmax (rows span 16-lane halves; reduce via shfl_xor) ---
    float scl[8];
#pragma unroll
    for (int r = 0; r < 8; ++r) {
      float v = s[0][r];
#pragma unroll
      for (int nk = 1; nk < 8; ++nk) v = fmaxf(v, s[nk][r]);
      v = fmaxf(v, __shfl_xor(v, 1, 16));
      v = fmaxf(v, __shfl_xor(v, 2, 16));
      v = fmaxf(v, __shfl_xor(v, 4, 16));
      v = fmaxf(v, __shfl_xor(v, 8, 16));
      float mn = fmaxf(rm[r], v);
      scl[r] = __expf(rm[r] - mn);
      rm[r] = mn;
    }
    float lc[8];
#pragma unroll
    for (int r = 0; r < 8; ++r) lc[r] = 0.0f;
#pragma unroll
    for (int nk = 0; nk < 8; ++nk)
#pragma unroll
      for (int r = 0; r < 8; ++r) {
        float p = __expf(s[nk][r] - rm[r]);
        s[nk][r] = p;
        lc[r] += p;
      }
#pragma unroll
    for (int r = 0; r < 8; ++r) {
      float v = lc[r];
      v += __shfl_xor(v, 1, 16);
      v += __shfl_xor(v, 2, 16);
      v += __shfl_xor(v, 4, 16);
      v += __shfl_xor(v, 8, 16);
      rl[r] = rl[r] * scl[r] + v;
    }
#pragma unroll
    for (int di = 0; di < 4; ++di)
#pragma unroll
      for (int r = 0; r < 8; ++r) o[di][r] *= scl[r];

    // --- stage P (C-layout -> A-layout via per-wave LDS; wave-local order) ---
#pragma unroll
    for (int nk = 0; nk < 8; ++nk)
#pragma unroll
      for (int r = 0; r < 8; ++r)
        Ps[w][r + hh * 8][nk * 16 + l16] = f2bf(s[nk][r]);

    // --- O += P * V ---
#pragma unroll
    for (int ks = 0; ks < 4; ++ks) {
      FragBF ap;
      const __bf16* pp = &Ps[w][l16][ks * 32 + bb];
      ap.h[0] = *(const v8bf*)pp;
      ap.h[1] = *(const v8bf*)(pp + 8);
#pragma unroll
      for (int di = 0; di < 4; ++di) {
        FragBF bv;
        const __bf16* vp = &Vs[cur][di * 16 + l16][ks * 32 + bb];
        bv.h[0] = *(const v8bf*)vp;
        bv.h[1] = *(const v8bf*)(vp + 8);
        o[di] = __builtin_amdgcn_wmma_f32_16x16x32_bf16(false, ap.v, false, bv.v,
                                                        (short)0, o[di], false, false);
      }
    }
  }

  // --- normalize + store Y (bf16, GEMM-A layout [t][NH*64]) ---
#pragma unroll
  for (int di = 0; di < 4; ++di)
#pragma unroll
    for (int r = 0; r < 8; ++r) {
      int m = q0 + w * 16 + r + hh * 8;
      int n = h * 64 + di * 16 + l16;
      Y[(size_t)m * (NH_ * HD_) + n] = f2bf(o[di][r] / rl[r]);
    }
}

// ---------------------------------------------------------------------------
// Host-side orchestration
// ---------------------------------------------------------------------------
extern "C" void kernel_launch(void* const* d_in, const int* in_sizes, int n_in,
                              void* d_out, int out_size, void* d_ws, size_t ws_size,
                              hipStream_t stream) {
  (void)in_sizes; (void)n_in; (void)out_size; (void)ws_size;

  const float* x    = (const float*)d_in[0];
  const float* amsk = (const float*)d_in[1];
  const float* wq   = (const float*)d_in[2];
  const float* wk   = (const float*)d_in[3];
  const float* wv   = (const float*)d_in[4];
  const float* wo   = (const float*)d_in[5];
  float* out = (float*)d_out;

  const int T = T_, DM = DM_;
  const int NQ = NH_ * HD_;    // 2048
  const int NK = NKV_ * HD_;   // 512

  // workspace carve-up (256B aligned)
  char* p = (char*)d_ws;
  auto alloc = [&](size_t bytes) {
    char* r = p;
    p += (bytes + 255) & ~(size_t)255;
    return r;
  };
  __bf16* xb  = (__bf16*)alloc((size_t)T * DM * 2);
  __bf16* wqT = (__bf16*)alloc((size_t)NQ * DM * 2);
  __bf16* wkT = (__bf16*)alloc((size_t)NK * DM * 2);
  __bf16* wvT = (__bf16*)alloc((size_t)NK * DM * 2);
  __bf16* woT = (__bf16*)alloc((size_t)DM * NQ * 2);
  float*  qf  = (float*)alloc((size_t)T * NQ * 4);
  float*  kf  = (float*)alloc((size_t)T * NK * 4);
  float*  vf  = (float*)alloc((size_t)T * NK * 4);
  __bf16* Qb  = (__bf16*)alloc((size_t)NH_ * T * HD_ * 2);
  __bf16* Kb  = (__bf16*)alloc((size_t)NKV_ * T * HD_ * 2);
  __bf16* Vt  = (__bf16*)alloc((size_t)NKV_ * HD_ * T * 2);
  __bf16* Y   = (__bf16*)alloc((size_t)T * NQ * 2);

  // 1) conversions
  {
    int n = T * DM;
    cvt_f32_bf16_kernel<<<(n + 255) / 256, 256, 0, stream>>>(x, xb, n);
  }
  transpose_f32_bf16_kernel<<<dim3(NQ / 32, DM / 32), dim3(32, 8), 0, stream>>>(wq, wqT, DM, NQ);
  transpose_f32_bf16_kernel<<<dim3(NK / 32, DM / 32), dim3(32, 8), 0, stream>>>(wk, wkT, DM, NK);
  transpose_f32_bf16_kernel<<<dim3(NK / 32, DM / 32), dim3(32, 8), 0, stream>>>(wv, wvT, DM, NK);
  transpose_f32_bf16_kernel<<<dim3(NQ / 32, DM / 32), dim3(32, 8), 0, stream>>>(wo, woT, NQ, DM);

  // 2) QKV projections (WMMA)
  gemm_bf16_nt_kernel<<<dim3(NQ / 128, T / 128), 128, 0, stream>>>(xb, wqT, qf, T, NQ, DM);
  gemm_bf16_nt_kernel<<<dim3(NK / 128, T / 128), 128, 0, stream>>>(xb, wkT, kf, T, NK, DM);
  gemm_bf16_nt_kernel<<<dim3(NK / 128, T / 128), 128, 0, stream>>>(xb, wvT, vf, T, NK, DM);

  // 3) RoPE + repack
  {
    int n = NH_ * T * 32;
    rope_pack_kernel<<<(n + 255) / 256, 256, 0, stream>>>(qf, Qb, T, NH_);
  }
  {
    int n = NKV_ * T * 32;
    rope_pack_kernel<<<(n + 255) / 256, 256, 0, stream>>>(kf, Kb, T, NKV_);
  }
  {
    int n = NKV_ * HD_ * T;
    pack_vT_kernel<<<(n + 255) / 256, 256, 0, stream>>>(vf, Vt, T);
  }

  // 4) flash attention (WMMA)
  flash_attn_kernel<<<dim3(T / 64, NH_), 128, 0, stream>>>(Qb, Kb, Vt, amsk, Y, T);

  // 5) output projection (WMMA) -> f32 out
  gemm_bf16_nt_kernel<<<dim3(DM / 128, T / 128), 128, 0, stream>>>(Y, woT, out, T, DM, NQ);
}
